// TrainedGCNModel_57011395887501
// MI455X (gfx1250) — matrix operand order repaired and checked by reading, not compiled, not run on hardware
//
#include <hip/hip_runtime.h>
#include <hip/hip_bf16.h>

typedef __attribute__((ext_vector_type(16))) _Float16 v16h;
typedef __attribute__((ext_vector_type(8)))  _Float16 v8h;
typedef __attribute__((ext_vector_type(8)))  float    v8f;

#define S_SUB 4096
#define G_GR  1024
#define H_DIM 128
#define D_DIM 64
#define BN_EPS 1e-5f

// ---------------- utility ----------------
__global__ void fill_f32(float* __restrict__ p, float v, int n) {
    int t = blockIdx.x * blockDim.x + threadIdx.x;
    if (t < n) p[t] = v;
}

// ---------------- substructure attention ----------------
__global__ void subst_accum(const float* __restrict__ x, float* __restrict__ sums,
                            float* __restrict__ cnt, int n) {
    int t = blockIdx.x * blockDim.x + threadIdx.x;
    int i = t >> 6, d = t & 63;
    if (i >= n) return;
    int sid = (int)x[i * D_DIM + 5];
    sid = sid < 0 ? 0 : (sid >= S_SUB ? S_SUB - 1 : sid);
    atomicAdd(&sums[sid * D_DIM + d], x[i * D_DIM + d]);
    if (d == 0) atomicAdd(&cnt[sid], 1.0f);
}

__global__ void subst_score(const float* __restrict__ sums, const float* __restrict__ cnt,
                            const float* __restrict__ w1, const float* __restrict__ b1,
                            const float* __restrict__ w2, float* __restrict__ scores) {
    int s = blockIdx.x * blockDim.x + threadIdx.x;
    if (s >= S_SUB) return;
    float inv = 1.0f / fmaxf(cnt[s], 1.0f);
    float m[D_DIM];
    for (int d = 0; d < D_DIM; ++d) m[d] = sums[s * D_DIM + d] * inv;
    float acc = 0.0f;
    for (int j = 0; j < 64; ++j) {
        float t = b1[j];
        for (int d = 0; d < D_DIM; ++d) t += m[d] * w1[d * 64 + j];
        acc += tanhf(t) * w2[j];
    }
    scores[s] = acc;
}

__global__ void softmax4096(const float* __restrict__ scores, float* __restrict__ attn) {
    __shared__ float red[1024];
    int t = threadIdx.x;
    float m = -1e30f;
    for (int s = t; s < S_SUB; s += 1024) m = fmaxf(m, scores[s]);
    red[t] = m; __syncthreads();
    for (int off = 512; off > 0; off >>= 1) {
        if (t < off) red[t] = fmaxf(red[t], red[t + off]);
        __syncthreads();
    }
    float mx = red[0]; __syncthreads();
    float sum = 0.0f;
    for (int s = t; s < S_SUB; s += 1024) sum += __expf(scores[s] - mx);
    red[t] = sum; __syncthreads();
    for (int off = 512; off > 0; off >>= 1) {
        if (t < off) red[t] += red[t + off];
        __syncthreads();
    }
    float tot = red[0];
    for (int s = t; s < S_SUB; s += 1024) attn[s] = __expf(scores[s] - mx) / tot;
}

// ---------------- feature assembly / weights ----------------
__global__ void build_h0(const float* __restrict__ x, const float* __restrict__ attn,
                         _Float16* __restrict__ h16, int n) {
    int t = blockIdx.x * blockDim.x + threadIdx.x;
    int i = t >> 7, k = t & 127;
    if (i >= n) return;
    float v;
    if (k < D_DIM) v = x[i * D_DIM + k];
    else if (k == D_DIM) {
        int sid = (int)x[i * D_DIM + 5];
        sid = sid < 0 ? 0 : (sid >= S_SUB ? S_SUB - 1 : sid);
        v = attn[sid];
    } else v = 0.0f;
    h16[t] = (_Float16)v;
}

// Wt layout: [layer][n=0..127][k=0..127], k contiguous (transposed weights)
__global__ void prep_w0(const float* __restrict__ cw0, _Float16* __restrict__ Wt) {
    int t = blockIdx.x * blockDim.x + threadIdx.x;   // 16384
    if (t >= H_DIM * H_DIM) return;
    int n = t >> 7, k = t & 127;
    Wt[t] = (k < D_DIM + 1) ? (_Float16)cw0[k * H_DIM + n] : (_Float16)0.0f;
}

__global__ void prep_wrest(const float* __restrict__ cwr, _Float16* __restrict__ Wt) {
    int t = blockIdx.x * blockDim.x + threadIdx.x;   // 3*16384
    if (t >= 3 * H_DIM * H_DIM) return;
    int layer = t >> 14, n = (t >> 7) & 127, k = t & 127;
    Wt[H_DIM * H_DIM + t] = (_Float16)cwr[layer * H_DIM * H_DIM + k * H_DIM + n];
}

// ---------------- degree / norm ----------------
__global__ void deg_accum(const int* __restrict__ col, float* __restrict__ deg, int E) {
    int e = blockIdx.x * blockDim.x + threadIdx.x;
    if (e < E) atomicAdd(&deg[col[e]], 1.0f);
}

__global__ void to_dinv(float* __restrict__ deg, int n) {
    int i = blockIdx.x * blockDim.x + threadIdx.x;
    if (i < n) deg[i] = rsqrtf(fmaxf(deg[i], 1.0f));
}

__global__ void gcnt_accum(const int* __restrict__ batch, float* __restrict__ gcnt, int n) {
    int i = blockIdx.x * blockDim.x + threadIdx.x;
    if (i < n) atomicAdd(&gcnt[batch[i]], 1.0f);
}

// ---------------- WMMA GEMM: C[N,128] = A16[N,128] x Wt[128n][128k]^T ----------------
__global__ __launch_bounds__(256) void gemm_wmma(const _Float16* __restrict__ A,
                                                 const _Float16* __restrict__ Wt,
                                                 float* __restrict__ C, int n_rows) {
    const int wave  = threadIdx.x >> 5;          // 0..7 : output column tile
    const int lane  = threadIdx.x & 31;
    const int khalf = lane >> 4;                 // 0 or 1 (K-half selector)
    const int lrow  = lane & 15;
    const int tile0 = blockIdx.x * 16;
    int row  = tile0 + lrow;
    int srow = row < n_rows ? row : (n_rows - 1);   // clamp: keep EXEC all-1 at WMMA
    const int ncol = wave * 16 + lrow;

    v8f acc = {};
    const _Float16* arow = A + (size_t)srow * H_DIM;
    const _Float16* brow = Wt + (size_t)ncol * H_DIM;

    #pragma unroll
    for (int kk = 0; kk < 4; ++kk) {
        // A 16x32 f16 tile: lane holds halves 0..7 = K[kk*32+khalf*8 ..], 8..15 = K[+16..]
        v8h alo = *(const v8h*)(arow + kk * 32 + khalf * 8);
        v8h ahi = *(const v8h*)(arow + kk * 32 + 16 + khalf * 8);
        v16h a;
        #pragma unroll
        for (int j = 0; j < 8; ++j) { a[j] = alo[j]; a[8 + j] = ahi[j]; }
        // B 32x16 f16 tile: lane(col=lrow) holds halves 0..7 = K[kk*32+khalf*8 ..], 8..15 = K[+16..]
        v8h blo = *(const v8h*)(brow + kk * 32 + khalf * 8);
        v8h bhi = *(const v8h*)(brow + kk * 32 + 16 + khalf * 8);
        v16h b;
        #pragma unroll
        for (int j = 0; j < 8; ++j) { b[j] = blo[j]; b[8 + j] = bhi[j]; }
        acc = __builtin_amdgcn_wmma_f32_16x16x32_f16(false, a, false, b,
                                                     (short)0, acc, false, false);
    }
    // C/D layout: acc[r] -> row = r + 8*khalf, col = ncol
    float* cbase = C + (size_t)(tile0 + 8 * khalf) * H_DIM + ncol;
    if (tile0 + 16 <= n_rows) {
        // full tile: wave-uniform branch -> straight-line stores, no EXEC churn
        #pragma unroll
        for (int r = 0; r < 8; ++r) cbase[(size_t)r * H_DIM] = acc[r];
    } else {
        #pragma unroll
        for (int r = 0; r < 8; ++r)
            if (tile0 + 8 * khalf + r < n_rows) cbase[(size_t)r * H_DIM] = acc[r];
    }
}

// ---------------- edge scatter: one wave per edge, lane = float4 chunk ----------------
__global__ void scatter_edges(const float* __restrict__ src,
                              const int* __restrict__ row, const int* __restrict__ col,
                              const float* __restrict__ dinv, float* __restrict__ agg,
                              int E, int n) {
    int t = blockIdx.x * blockDim.x + threadIdx.x;
    int e = t >> 5, f4 = t & 31;
    if (e >= E + n) return;
    int r, c;
    if (e < E) { r = row[e]; c = col[e]; } else { r = c = e - E; }
    float nrm = dinv[r] * dinv[c];
    float4 v = ((const float4*)(src + (size_t)r * H_DIM))[f4];
    float* dst = agg + (size_t)c * H_DIM + f4 * 4;
    atomicAdd(dst + 0, v.x * nrm);
    atomicAdd(dst + 1, v.y * nrm);
    atomicAdd(dst + 2, v.z * nrm);
    atomicAdd(dst + 3, v.w * nrm);
}

// ---------------- bias + BN + activation (+ optional mean-pool accumulate) ----------------
__global__ void post_bn_act(const float* __restrict__ agg, const float* __restrict__ bias,
                            const float* __restrict__ gamma, const float* __restrict__ beta,
                            const float* __restrict__ mean, const float* __restrict__ var,
                            _Float16* __restrict__ h16, float* __restrict__ pooled,
                            const int* __restrict__ batch, int n, int use_relu, int do_pool) {
    int t = blockIdx.x * blockDim.x + threadIdx.x;
    int i = t >> 7, f = t & 127;
    if (i >= n) return;
    float v = agg[t] + bias[f];
    v = (v - mean[f]) * rsqrtf(var[f] + BN_EPS) * gamma[f] + beta[f];
    v = use_relu ? fmaxf(v, 0.0f) : (v > 0.0f ? v : expm1f(v));
    h16[t] = (_Float16)v;
    if (do_pool) atomicAdd(&pooled[(size_t)batch[i] * H_DIM + f], v);
}

// ---------------- prediction head ----------------
__global__ void head_kernel(const float* __restrict__ pooled, const float* __restrict__ gcnt,
                            const float* __restrict__ hw1, const float* __restrict__ hb1,
                            const float* __restrict__ hw2, const float* __restrict__ hb2,
                            float* __restrict__ out) {
    int g = blockIdx.x * blockDim.x + threadIdx.x;
    if (g >= G_GR) return;
    float inv = 1.0f / fmaxf(gcnt[g], 1.0f);
    float p[H_DIM];
    for (int f = 0; f < H_DIM; ++f) p[f] = pooled[g * H_DIM + f] * inv;
    float acc = hb2[0];
    for (int j = 0; j < 64; ++j) {
        float s = hb1[j];
        for (int f = 0; f < H_DIM; ++f) s += p[f] * hw1[f * 64 + j];
        s = s > 0.0f ? s : expm1f(s);
        acc += s * hw2[j];
    }
    out[g] = acc;
}

// ---------------- host orchestration ----------------
static inline size_t alignUp(size_t v) { return (v + 255) & ~(size_t)255; }

extern "C" void kernel_launch(void* const* d_in, const int* in_sizes, int n_in,
                              void* d_out, int out_size, void* d_ws, size_t ws_size,
                              hipStream_t stream) {
    const float* x        = (const float*)d_in[0];
    const int*   eidx     = (const int*)d_in[1];
    const int*   batch    = (const int*)d_in[2];
    const float* w1       = (const float*)d_in[3];
    const float* b1       = (const float*)d_in[4];
    const float* w2       = (const float*)d_in[5];
    const float* cw0      = (const float*)d_in[6];
    const float* cb0      = (const float*)d_in[7];
    const float* cw_rest  = (const float*)d_in[8];
    const float* cb_rest  = (const float*)d_in[9];
    const float* bn_gamma = (const float*)d_in[10];
    const float* bn_beta  = (const float*)d_in[11];
    const float* bn_mean  = (const float*)d_in[12];
    const float* bn_var   = (const float*)d_in[13];
    const float* hw1      = (const float*)d_in[14];
    const float* hb1      = (const float*)d_in[15];
    const float* hw2      = (const float*)d_in[16];
    const float* hb2      = (const float*)d_in[17];
    float* out = (float*)d_out;

    const int N = in_sizes[2];
    const int E = in_sizes[1] / 2;
    const int* row = eidx;
    const int* col = eidx + E;

    // workspace carve-up
    char* ws = (char*)d_ws;
    size_t off = 0;
    auto take = [&](size_t bytes) { char* p = ws + off; off += alignUp(bytes); return p; };
    float*    subst_cnt  = (float*)take(S_SUB * 4);
    float*    subst_sums = (float*)take((size_t)S_SUB * D_DIM * 4);
    float*    scores     = (float*)take(S_SUB * 4);
    float*    attn       = (float*)take(S_SUB * 4);
    float*    dinv       = (float*)take((size_t)N * 4);
    float*    gcnt       = (float*)take(G_GR * 4);
    float*    pooled     = (float*)take((size_t)G_GR * H_DIM * 4);
    _Float16* Wt         = (_Float16*)take((size_t)4 * H_DIM * H_DIM * 2);
    _Float16* h16        = (_Float16*)take((size_t)N * H_DIM * 2);
    float*    gemm_out   = (float*)take((size_t)N * H_DIM * 4);
    float*    agg        = (float*)take((size_t)N * H_DIM * 4);
    (void)ws_size; (void)n_in; (void)out_size;

    const int TB = 256;
    auto blocks = [](long long n, int tb) { return (unsigned)((n + tb - 1) / tb); };

    // phase 0: substructure attention
    fill_f32<<<blocks(S_SUB, TB), TB, 0, stream>>>(subst_cnt, 0.0f, S_SUB);
    fill_f32<<<blocks((long long)S_SUB * D_DIM, TB), TB, 0, stream>>>(subst_sums, 0.0f, S_SUB * D_DIM);
    subst_accum<<<blocks((long long)N * D_DIM, TB), TB, 0, stream>>>(x, subst_sums, subst_cnt, N);
    subst_score<<<blocks(S_SUB, TB), TB, 0, stream>>>(subst_sums, subst_cnt, w1, b1, w2, scores);
    softmax4096<<<1, 1024, 0, stream>>>(scores, attn);

    // phase 1: features, degrees, weights, pool counters
    build_h0<<<blocks((long long)N * H_DIM, TB), TB, 0, stream>>>(x, attn, h16, N);
    fill_f32<<<blocks(N, TB), TB, 0, stream>>>(dinv, 1.0f, N);   // self-loop contribution
    deg_accum<<<blocks(E, TB), TB, 0, stream>>>(col, dinv, E);
    to_dinv<<<blocks(N, TB), TB, 0, stream>>>(dinv, N);
    prep_w0<<<blocks(H_DIM * H_DIM, TB), TB, 0, stream>>>(cw0, Wt);
    prep_wrest<<<blocks(3 * H_DIM * H_DIM, TB), TB, 0, stream>>>(cw_rest, Wt);
    fill_f32<<<blocks((long long)G_GR * H_DIM, TB), TB, 0, stream>>>(pooled, 0.0f, G_GR * H_DIM);
    fill_f32<<<blocks(G_GR, TB), TB, 0, stream>>>(gcnt, 0.0f, G_GR);
    gcnt_accum<<<blocks(N, TB), TB, 0, stream>>>(batch, gcnt, N);

    // phase 2: four GCN layers
    for (int l = 0; l < 4; ++l) {
        gemm_wmma<<<blocks(N, 16), TB, 0, stream>>>(h16, Wt + (size_t)l * H_DIM * H_DIM,
                                                    gemm_out, N);
        fill_f32<<<blocks((long long)N * H_DIM, TB), TB, 0, stream>>>(agg, 0.0f, N * H_DIM);
        scatter_edges<<<blocks((long long)(E + N) * 32, TB), TB, 0, stream>>>(
            gemm_out, row, col, dinv, agg, E, N);
        const float* bias = (l == 0) ? cb0 : (cb_rest + (size_t)(l - 1) * H_DIM);
        post_bn_act<<<blocks((long long)N * H_DIM, TB), TB, 0, stream>>>(
            agg, bias, bn_gamma + (size_t)l * H_DIM, bn_beta + (size_t)l * H_DIM,
            bn_mean + (size_t)l * H_DIM, bn_var + (size_t)l * H_DIM,
            h16, pooled, batch, N, /*use_relu=*/(l == 3), /*do_pool=*/(l == 3));
    }

    // phase 3: head
    head_kernel<<<blocks(G_GR, TB), TB, 0, stream>>>(pooled, gcnt, hw1, hb1, hw2, hb2, out);
}
